// Encoder_56770877718765
// MI455X (gfx1250) — matrix operand order repaired and checked by reading, not compiled, not run on hardware
//
#include <hip/hip_runtime.h>
#include <hip/hip_bf16.h>

#define T_STEPS 512
#define BATCH   256
#define HID     256
#define G4H     1024   // 4*HID
#define VOCAB   32000
#define NWG     32     // persistent workgroups (must all be co-resident)
#define WLDS    262144 // per-WG weight-fragment LDS (256 KB of 320 KB)

typedef __attribute__((ext_vector_type(8)))  float  v8f;
typedef __attribute__((ext_vector_type(8)))  __bf16 v8bf;
typedef __attribute__((ext_vector_type(16))) __bf16 v16bf;

__device__ __forceinline__ unsigned short f2bf(float f) {
  unsigned int u = __float_as_uint(f);
  u += 0x7FFFu + ((u >> 16) & 1u);            // round-to-nearest-even
  return (unsigned short)(u >> 16);
}

// CDNA5 has a hardware transcendental TANH -> 1 TRANS op per activation.
#if __has_builtin(__builtin_amdgcn_tanhf)
__device__ __forceinline__ float htanh(float x) { return __builtin_amdgcn_tanhf(x); }
#else
__device__ __forceinline__ float htanh(float x) { return 2.0f / (1.0f + __expf(-2.0f * x)) - 1.0f; }
#endif
__device__ __forceinline__ float hsig(float x)  { return 0.5f + 0.5f * htanh(0.5f * x); }

// Device-wide barrier: monotonic counter (no ABA), reset each launch by prep.
__device__ __forceinline__ void grid_barrier(unsigned int* cnt, unsigned int target) {
  __syncthreads();
  if (threadIdx.x == 0) {
    __threadfence();                              // flush h-state stores device-wide
    atomicAdd(cnt, 1u);
    while (__hip_atomic_load(cnt, __ATOMIC_RELAXED, __HIP_MEMORY_SCOPE_AGENT) < target)
      __builtin_amdgcn_s_sleep(1);
    __threadfence();                              // acquire others' stores
  }
  __syncthreads();
}

// ---------------------------------------------------------------------------
// Prep 1: embedding table f32 -> bf16, once.
// ---------------------------------------------------------------------------
__global__ void lstm_prep_embf(const float* __restrict__ embed,
                               unsigned short* __restrict__ embf) {
  int idx = (blockIdx.x * blockDim.x + threadIdx.x) * 4;
  if (idx < VOCAB * HID) {
    float4 f = *(const float4*)(embed + idx);
    ushort4 o;
    o.x = f2bf(f.x); o.y = f2bf(f.y); o.z = f2bf(f.z); o.w = f2bf(f.w);
    *(ushort4*)(embf + idx) = o;
  }
}

// ---------------------------------------------------------------------------
// Prep 2: swizzle W_ih|W_hh (f32) into WMMA B-fragment order (bf16).
// Layout: [jt(4)][kc(16)][subtile(16)][lane(32)][32B]
// ---------------------------------------------------------------------------
__global__ void lstm_prep_bfrag(const float* __restrict__ Wih,
                                const float* __restrict__ Whh,
                                unsigned int* __restrict__ bfrag) {
  const int s    = blockIdx.x * blockDim.x + threadIdx.x;   // 0..32767 lane-slots
  const int lane = s & 31;
  const int sub  = (s >> 5) & 15;
  const int kc   = (s >> 9) & 15;
  const int jt   = (s >> 13) & 3;

  const int n_in  = lane & 15;
  const int khalf = lane >> 4;
  const int n     = sub * 16 + n_in;
  const int g     = n >> 6;
  const int jj    = n & 63;
  const int w_row = g * HID + jt * 64 + jj;
  const int kbase = kc * 32 + khalf * 8;

  const float* src = (kbase < HID)
      ? (Wih + (size_t)w_row * HID + kbase)
      : (Whh + (size_t)w_row * HID + (kbase - HID));

  float4 f0 = *(const float4*)(src + 0);
  float4 f1 = *(const float4*)(src + 4);
  float4 f2 = *(const float4*)(src + 16);
  float4 f3 = *(const float4*)(src + 20);
  float fv[16] = {f0.x,f0.y,f0.z,f0.w, f1.x,f1.y,f1.z,f1.w,
                  f2.x,f2.y,f2.z,f2.w, f3.x,f3.y,f3.z,f3.w};
  alignas(16) unsigned int out[8];
  #pragma unroll
  for (int v = 0; v < 8; ++v)
    out[v] = (unsigned int)f2bf(fv[2*v]) | ((unsigned int)f2bf(fv[2*v+1]) << 16);

  uint4* dst = (uint4*)(bfrag + (size_t)s * 8);   // 32B per slot
  dst[0] = *(const uint4*)(out + 0);
  dst[1] = *(const uint4*)(out + 4);
}

// ---------------- Prep 3: bias_sum, h0 -> bf16, barrier reset ----------------
__global__ void lstm_prep_misc(const float* __restrict__ b_ih,
                               const float* __restrict__ b_hh,
                               float* __restrict__ bias,
                               const float* __restrict__ h0,
                               unsigned short* __restrict__ hbf0,
                               unsigned int* __restrict__ barcnt) {
  int idx = blockIdx.x * blockDim.x + threadIdx.x;
  if (idx == 0) *barcnt = 0;                       // fresh barrier each launch
  if (idx < G4H) bias[idx] = b_ih[idx] + b_hh[idx];
  if (idx < BATCH * HID) hbf0[idx] = f2bf(h0[idx]);
}

// ---------------------------------------------------------------------------
// Persistent LSTM: 32 WGs x 256 threads run ALL 512 timesteps.
// Weights prefilled into LDS once (256 KB/WG), read via ds_load_b128 with
// one-iteration-ahead double buffering; A fragments (token gather + h state)
// prefetched from global one k-chunk ahead. c-state stays in registers for
// the whole sequence; h exchanged via double-buffered bf16 global buffer
// + one device-wide barrier per step. An empty asm() re-defines the LDS
// offset each step so LICM cannot hoist fragment loads (scratch spills).
// ---------------------------------------------------------------------------
__global__ __launch_bounds__(256)
void lstm_persist(const int* __restrict__ tokens,
                  const unsigned short* __restrict__ embf,
                  const unsigned int* __restrict__ bfrag,
                  const float* __restrict__ bias,
                  const float* __restrict__ c0,
                  float* __restrict__ h_out, float* __restrict__ c_out,
                  unsigned short* __restrict__ hbf0,
                  unsigned short* __restrict__ hbf1,
                  unsigned int* __restrict__ barcnt)
{
  extern __shared__ char smem[];                  // WLDS bytes of weight fragments

  const int tid   = threadIdx.x;
  const int lane  = tid & 31;
  const int wave  = tid >> 5;
  const int m_sub = wave & 1;       // 16-row strip within 32-row tile
  const int jq    = wave >> 1;      // 16-hidden-unit subtile within 64

  const int bm = (blockIdx.x & 7) * 32;   // batch tile base
  const int jt = blockIdx.x >> 3;         // hidden tile index
  const int j0 = jt * 64;

  // ---- one-time prefill: this WG's weight fragments -> LDS ----
  {
    const char* gw = (const char*)bfrag + (size_t)jt * 262144;
    #pragma unroll 4
    for (int i = 0; i < 64; ++i) {
      *(uint4*)(smem + i * 4096 + tid * 16) =
          *(const uint4*)(gw + i * 4096 + tid * 16);
    }
  }
  __syncthreads();

  const int n_in  = lane & 15;
  const int khalf = (lane >> 4) & 1;
  const int b_row = bm + m_sub * 16 + n_in;

  const unsigned int boff = (unsigned int)(jq * 1024 + lane * 32);

  // per-lane constants for the epilogue
  const int jj = jq * 16 + n_in;
  const int j  = j0 + jj;
  const float bi = bias[          j];
  const float bf = bias[  HID   + j];
  const float bg = bias[2*HID   + j];
  const float bo = bias[3*HID   + j];
  const int mbase = bm + m_sub * 16 + khalf * 8;   // C/D: lanes 16-31 hold M+8

  // c-state in registers for the entire sequence
  float creg[8];
  #pragma unroll
  for (int v = 0; v < 8; ++v)
    creg[v] = c0[(size_t)(mbase + v) * HID + j];

  const unsigned short* hb_rd[2] = { hbf0 + (size_t)b_row * HID + khalf * 8,
                                     hbf1 + (size_t)b_row * HID + khalf * 8 };
  unsigned short* hb_wr[2] = { hbf1, hbf0 };

  const v8f vzero = {0.f,0.f,0.f,0.f,0.f,0.f,0.f,0.f};
  union ABu { v16bf v; v8bf h[2]; };

  for (int t = 0; t < T_STEPS; ++t) {
    const unsigned short* xb = embf + (size_t)tokens[t * BATCH + b_row] * HID + khalf * 8;
    const unsigned short* hb = hb_rd[t & 1];
    unsigned short*       hw = hb_wr[t & 1];

    unsigned int bo2 = boff;
    asm volatile("" : "+v"(bo2));        // opaque per-step: blocks LICM hoisting
    const char* bl = smem + bo2;

    v8f acc[4];
    #pragma unroll
    for (int g = 0; g < 4; ++g) acc[g] = vzero;

    ABu a[2];
    ABu bfr[2][4];

    // prologue: fragments for kc = 0
    a[0].h[0] = *(const v8bf*)(xb);
    a[0].h[1] = *(const v8bf*)(xb + 16);
    #pragma unroll
    for (int g = 0; g < 4; ++g) {
      const char* p = bl + g * 4096;
      bfr[0][g].h[0] = *(const v8bf*)(p);
      bfr[0][g].h[1] = *(const v8bf*)(p + 16);
    }

    #pragma unroll
    for (int kc = 0; kc < 16; ++kc) {
      const int cur = kc & 1;
      const int nxt = cur ^ 1;
      if (kc + 1 < 16) {                 // issue kc+1's loads before kc's WMMAs
        const int kn = kc + 1;
        const unsigned short* base = (kn < 8) ? (xb + kn * 32) : (hb + (kn - 8) * 32);
        a[nxt].h[0] = *(const v8bf*)(base);
        a[nxt].h[1] = *(const v8bf*)(base + 16);
        #pragma unroll
        for (int g = 0; g < 4; ++g) {
          const char* p = bl + kn * 16384 + g * 4096;   // subtile = g*4 + jq
          bfr[nxt][g].h[0] = *(const v8bf*)(p);
          bfr[nxt][g].h[1] = *(const v8bf*)(p + 16);
        }
      }
      #pragma unroll
      for (int g = 0; g < 4; ++g)
        acc[g] = __builtin_amdgcn_wmma_f32_16x16x32_bf16(
            false, a[cur].v, false, bfr[cur][g].v, (short)0, acc[g], false, false);
    }

    // -------- in-register LSTM epilogue (c never leaves VGPRs) --------
    #pragma unroll
    for (int v = 0; v < 8; ++v) {
      const size_t idx = (size_t)(mbase + v) * HID + j;
      float gi = acc[0][v] + bi;
      float gf = acc[1][v] + bf;
      float gg = acc[2][v] + bg;
      float go = acc[3][v] + bo;
      float ig = hsig(gi);
      float fg = hsig(gf);
      float cg = htanh(gg);
      float og = hsig(go);
      float c_new = fg * creg[v] + ig * cg;
      float h_new = og * htanh(c_new);
      creg[v] = c_new;
      hw[idx] = f2bf(h_new);             // next step's A-fragment source
      if (t == T_STEPS - 1) {            // final outputs only
        h_out[idx] = h_new;
        c_out[idx] = c_new;
      }
    }

    if (t + 1 < T_STEPS)
      grid_barrier(barcnt, (unsigned int)(t + 1) * NWG);
  }
}

extern "C" void kernel_launch(void* const* d_in, const int* in_sizes, int n_in,
                              void* d_out, int out_size, void* d_ws, size_t ws_size,
                              hipStream_t stream) {
  (void)in_sizes; (void)n_in; (void)out_size; (void)ws_size;

  const int*   tokens = (const int*)  d_in[0];  // [T,B] int32
  const float* h0     = (const float*)d_in[1];  // [B,H]
  const float* c0     = (const float*)d_in[2];  // [B,H]
  const float* embed  = (const float*)d_in[3];  // [V,H]
  const float* W_ih   = (const float*)d_in[4];  // [4H,H]
  const float* W_hh   = (const float*)d_in[5];  // [4H,H]
  const float* b_ih   = (const float*)d_in[6];  // [4H]
  const float* b_hh   = (const float*)d_in[7];  // [4H]

  float* h_out = (float*)d_out;                 // final h
  float* c_out = h_out + BATCH * HID;           // final c

  // workspace layout (~17.7 MB total)
  char* ws = (char*)d_ws;
  unsigned short* embf   = (unsigned short*)(ws);                  // 16,384,000 B
  unsigned int*   bfrag  = (unsigned int*)  (ws + 16384000);       //  1,048,576 B
  float*          bias   = (float*)         (ws + 17432576);       //      4,096 B
  unsigned short* hbf0   = (unsigned short*)(ws + 17436672);       //    131,072 B
  unsigned short* hbf1   = (unsigned short*)(ws + 17567744);       //    131,072 B
  unsigned int*   barcnt = (unsigned int*)  (ws + 17698816);       //          4 B

  lstm_prep_embf<<<(VOCAB * HID / 4 + 255) / 256, 256, 0, stream>>>(embed, embf);
  lstm_prep_bfrag<<<128, 256, 0, stream>>>(W_ih, W_hh, bfrag);
  lstm_prep_misc<<<(BATCH * HID + 255) / 256, 256, 0, stream>>>(b_ih, b_hh, bias,
                                                                h0, hbf0, barcnt);

  lstm_persist<<<NWG, 256, WLDS, stream>>>(tokens, embf, bfrag, bias, c0,
                                           h_out, c_out, hbf0, hbf1, barcnt);
}
